// MambaEncoder_79748952752196
// MI455X (gfx1250) — compile-verified, matrix operand order checked by reading, not run on hardware
//
#include <hip/hip_runtime.h>

#define DM     1024                 // D_MODEL
#define DI     2048                 // D_INNER
#define DSZ    16                   // D_STATE
#define DTR    64                   // DT_RANK
#define NDEPTH 4
#define NB     4
#define NSEQ   512
#define NROWS  (NB * NSEQ)          // 2048
#define NXC    (DTR + 2 * DSZ)      // 96

typedef unsigned short u16;
typedef __attribute__((ext_vector_type(16))) __bf16 bf16x16;
typedef __attribute__((ext_vector_type(8)))  float  f32x8;
typedef __attribute__((ext_vector_type(4)))  unsigned int u32x4;

union ABPack { u32x4 q[2]; bf16x16 v; };

__device__ __forceinline__ u16 f2bf(float f) {
  unsigned u = __float_as_uint(f);
  u += 0x7FFFu + ((u >> 16) & 1u);          // round-to-nearest-even
  return (u16)(u >> 16);
}
__device__ __forceinline__ float softplusf(float x) {
  return (x > 20.f) ? x : log1pf(__expf(x));
}
__device__ __forceinline__ float siluf(float x) {
  return x / (1.f + __expf(-x));
}

// ---------------------------------------------------------------------------
// Masked copy of input into the residual stream (d_out).
// ---------------------------------------------------------------------------
__global__ void mamba_mask_copy(const float* __restrict__ x,
                                const unsigned char* __restrict__ mask,
                                float* __restrict__ out, int total) {
  int i = blockIdx.x * blockDim.x + threadIdx.x;
  if (i >= total) return;
  int row = i / DM;
  out[i] = mask[row] ? x[i] : 0.f;
}

// ---------------------------------------------------------------------------
// Pack a K x N fp32 row-major weight into the bf16 per-lane WMMA B layout:
// flat index (((k/32)*N + n)*2 + khalf)*16 + j, where lane(khalf) holds
// j<8 -> k = 32*kt + khalf*8 + j ; j>=8 -> k = 32*kt + 16 + khalf*8 + (j-8).
// Each lane then reads its 16 bf16 B elements as two contiguous 16B loads.
// ---------------------------------------------------------------------------
__global__ void mamba_pack_w(const float* __restrict__ W, u16* __restrict__ P,
                             int K, int N) {
  size_t i = (size_t)blockIdx.x * blockDim.x + threadIdx.x;
  if (i >= (size_t)K * N) return;
  int k = (int)(i / N), n = (int)(i % N);
  int kt = k >> 5, kk = k & 31;
  int hi = (kk >> 4) & 1;
  int r  = kk & 15;
  int kh = r >> 3;
  int j  = (hi ? 8 : 0) + (r & 7);
  P[(((size_t)kt * N + n) * 2 + kh) * 16 + j] = f2bf(W[i]);
}

// ---------------------------------------------------------------------------
// RMSNorm over rows of 1024, output bf16 for the WMMA A operand.
// ---------------------------------------------------------------------------
__global__ __launch_bounds__(256)
void mamba_rmsnorm_bf16(const float* __restrict__ x, const float* __restrict__ w,
                        u16* __restrict__ out) {
  __shared__ float red[8];
  __shared__ float rinv;
  const int row = blockIdx.x;
  const float* xr = x + (size_t)row * DM;
  float s = 0.f;
  for (int c = threadIdx.x; c < DM; c += 256) { float v = xr[c]; s += v * v; }
  for (int o = 16; o > 0; o >>= 1) s += __shfl_xor(s, o, 32);
  if ((threadIdx.x & 31) == 0) red[threadIdx.x >> 5] = s;
  __syncthreads();
  if (threadIdx.x == 0) {
    float t = 0.f;
    for (int i = 0; i < 8; ++i) t += red[i];
    rinv = rsqrtf(t / (float)DM + 1e-5f);
  }
  __syncthreads();
  const float r = rinv;
  for (int c = threadIdx.x; c < DM; c += 256)
    out[(size_t)row * DM + c] = f2bf(xr[c] * r * w[c]);
}

// ---------------------------------------------------------------------------
// bf16 WMMA GEMM with register double-buffering: C(M,N) = A(M,K) * B(K,N).
// A row-major bf16, B packed by mamba_pack_w, C fp32.
// Block = 8 waves = 128x64 tile; each wave = 64x16 (4 stacked accumulators,
// one B fragment reused 4x).  The K loop is hand-unrolled by 2 (all K/32 are
// even here) with two static fragment buffers so the 10 global_load_b128 for
// step k+1 are in flight while the 4 v_wmma_f32_16x16x32_bf16 of step k
// execute — waits then cover loads issued a full MMA-group earlier.
// EPI 0: plain store.  EPI 1: softplus(acc + bias[col]).
// EPI 2: C = (C + acc) * mask[row]   (residual + mask, in-place).
// ---------------------------------------------------------------------------
template <int EPI>
__global__ __launch_bounds__(256)
void mamba_gemm_wmma(const u16* __restrict__ A, const u16* __restrict__ Bp,
                     float* __restrict__ C, int M, int N, int K,
                     const float* __restrict__ bias,
                     const unsigned char* __restrict__ mask) {
  const int lane   = threadIdx.x & 31;
  const int wave   = threadIdx.x >> 5;
  const int wm     = wave >> 2;                 // 0..1  (M)
  const int wn     = wave & 3;                  // 0..3  (N)
  const int tile_n = blockIdx.x * 4 + wn;
  if (tile_n * 16 >= N) return;                 // wave-uniform exit
  const int row0  = blockIdx.y * 128 + wm * 64;
  const int lrow  = lane & 15;
  const int khalf = lane >> 4;
  const int col   = tile_n * 16;

  f32x8 acc[4];
  for (int mi = 0; mi < 4; ++mi)
    for (int i = 0; i < 8; ++i) acc[mi][i] = 0.f;

  const int ksteps = K >> 5;                    // even for all K used here
  const u16* bbase = Bp + ((size_t)(col + lrow) * 2 + khalf) * 16;
  const u16* abase[4];
#pragma unroll
  for (int mi = 0; mi < 4; ++mi)
    abase[mi] = A + (size_t)(row0 + mi * 16 + lrow) * K + khalf * 8;

  ABPack afrag[2][4];
  ABPack bfrag[2];

  auto load_step = [&](int kt, ABPack (&a)[4], ABPack& b) {
    const u32x4* bptr = (const u32x4*)(bbase + (size_t)kt * N * 32);
    b.q[0] = bptr[0];
    b.q[1] = bptr[1];
#pragma unroll
    for (int mi = 0; mi < 4; ++mi) {
      const u16* arow = abase[mi] + kt * 32;
      a[mi].q[0] = *(const u32x4*)arow;             // k = khalf*8 .. +7
      a[mi].q[1] = *(const u32x4*)(arow + 16);      // k = 16+khalf*8 .. +7
    }
  };
  auto mma_step = [&](ABPack (&a)[4], ABPack& b) {
#pragma unroll
    for (int mi = 0; mi < 4; ++mi)
      acc[mi] = __builtin_amdgcn_wmma_f32_16x16x32_bf16(
          false, a[mi].v, false, b.v, (short)0, acc[mi], false, false);
  };

  load_step(0, afrag[0], bfrag[0]);
  for (int kt = 0; kt < ksteps; kt += 2) {
    load_step(kt + 1, afrag[1], bfrag[1]);          // ksteps even -> in range
    if (kt + 2 < ksteps)                            // -> global_prefetch_b8
      __builtin_prefetch(bbase + (size_t)(kt + 2) * N * 32, 0, 1);
    mma_step(afrag[0], bfrag[0]);
    if (kt + 2 < ksteps)
      load_step(kt + 2, afrag[0], bfrag[0]);
    mma_step(afrag[1], bfrag[1]);
  }

  const int rsel = (lane < 16) ? 0 : 8;         // C/D layout: VGPR v -> M=v / v+8
  const int cc   = col + lrow;
  for (int mi = 0; mi < 4; ++mi) {
    const int rb = row0 + mi * 16 + rsel;
#pragma unroll
    for (int v = 0; v < 8; ++v) {
      const int r = rb + v;
      float val = acc[mi][v];
      size_t idx = (size_t)r * N + cc;
      if (EPI == 0) {
        C[idx] = val;
      } else if (EPI == 1) {
        C[idx] = softplusf(val + bias[cc]);
      } else {
        float m = mask[r] ? 1.f : 0.f;
        C[idx] = (C[idx] + val) * m;
      }
    }
  }
}

// ---------------------------------------------------------------------------
// Depthwise causal conv(4) + bias + SiLU.  Reads u = xz[:, :DI], writes both
// fp32 (for the scan) and bf16 (A operand of x_proj GEMM).
// ---------------------------------------------------------------------------
__global__ void mamba_conv_silu(const float* __restrict__ xz,
                                const float* __restrict__ cw,
                                const float* __restrict__ cb,
                                u16* __restrict__ ubf, float* __restrict__ uf) {
  int i = blockIdx.x * blockDim.x + threadIdx.x;
  if (i >= NROWS * DI) return;
  int d = i & (DI - 1);
  int t = (i >> 11) & (NSEQ - 1);
  int b = i >> 20;
  float acc = cb[d];
#pragma unroll
  for (int j = 0; j < 4; ++j) {
    int tt = t + j - 3;
    if (tt >= 0)
      acc += xz[(size_t)(b * NSEQ + tt) * (2 * DI) + d] * cw[d * 4 + j];
  }
  float s = siluf(acc);
  uf[i]  = s;
  ubf[i] = f2bf(s);
}

// dt-rank slice of xdbc -> bf16 (A operand of the dt GEMM)
__global__ void mamba_dtcvt(const float* __restrict__ xdbc, u16* __restrict__ dtbf) {
  int i = blockIdx.x * blockDim.x + threadIdx.x;
  if (i >= NROWS * DTR) return;
  int row = i >> 6, c = i & 63;
  dtbf[i] = f2bf(xdbc[(size_t)row * NXC + c]);
}

// ---------------------------------------------------------------------------
// Selective scan: one thread per (b, d); 16-wide state in registers,
// per-timestep B/C (32 floats) broadcast through LDS. Fuses +u*D, *silu(z),
// and bf16 conversion for the out_proj A operand.
// ---------------------------------------------------------------------------
__global__ __launch_bounds__(256)
void mamba_scan(const float* __restrict__ dt, const float* __restrict__ xdbc,
                const float* __restrict__ uf, const float* __restrict__ xz,
                const float* __restrict__ A_log, const float* __restrict__ Dp,
                u16* __restrict__ ybf) {
  __shared__ float sBC[2 * DSZ];
  const int b = blockIdx.x >> 3;
  const int d = ((blockIdx.x & 7) << 8) + threadIdx.x;
  float Aneg[DSZ], h[DSZ];
#pragma unroll
  for (int n = 0; n < DSZ; ++n) {
    Aneg[n] = -__expf(A_log[(size_t)d * DSZ + n]);
    h[n] = 0.f;
  }
  const float Dv = Dp[d];
  for (int t = 0; t < NSEQ; ++t) {
    const size_t row = (size_t)b * NSEQ + t;
    if (threadIdx.x < 2 * DSZ)
      sBC[threadIdx.x] = xdbc[row * NXC + DTR + threadIdx.x];
    __syncthreads();
    const float dtv = dt[row * DI + d];
    const float uv  = uf[row * DI + d];
    const float zv  = xz[row * (2 * DI) + DI + d];
    const float du  = dtv * uv;
    float y = 0.f;
#pragma unroll
    for (int n = 0; n < DSZ; ++n) {
      float dA = __expf(dtv * Aneg[n]);
      h[n] = dA * h[n] + du * sBC[n];
      y += h[n] * sBC[DSZ + n];
    }
    y = (y + uv * Dv) * siluf(zv);
    ybf[row * DI + d] = f2bf(y);
    __syncthreads();
  }
}

// ---------------------------------------------------------------------------
extern "C" void kernel_launch(void* const* d_in, const int* in_sizes, int n_in,
                              void* d_out, int out_size, void* d_ws, size_t ws_size,
                              hipStream_t stream) {
  const float* x          = (const float*)d_in[0];
  const unsigned char* mk = (const unsigned char*)d_in[1];  // jnp.bool_ = 1 byte
  const float* in_proj_w  = (const float*)d_in[2];
  const float* conv_w     = (const float*)d_in[3];
  const float* conv_b     = (const float*)d_in[4];
  const float* x_proj_w   = (const float*)d_in[5];
  const float* dt_proj_w  = (const float*)d_in[6];
  const float* dt_proj_b  = (const float*)d_in[7];
  const float* A_log      = (const float*)d_in[8];
  const float* D_param    = (const float*)d_in[9];
  const float* out_proj_w = (const float*)d_in[10];
  const float* norm_w     = (const float*)d_in[11];
  float* out = (float*)d_out;

  char* ws = (char*)d_ws;
  size_t off = 0;
  auto carve = [&](size_t bytes) -> char* {
    off = (off + 255) & ~(size_t)255;
    char* p = ws + off;
    off += bytes;
    return p;
  };

  u16*   pk_in  = (u16*)carve((size_t)NDEPTH * DM * (2 * DI) * 2);
  u16*   pk_x   = (u16*)carve((size_t)NDEPTH * DI * NXC * 2);
  u16*   pk_dt  = (u16*)carve((size_t)NDEPTH * DTR * DI * 2);
  u16*   pk_out = (u16*)carve((size_t)NDEPTH * DI * DM * 2);
  u16*   hbf    = (u16*)carve((size_t)NROWS * DM * 2);
  float* xz     = (float*)carve((size_t)NROWS * 2 * DI * 4);
  u16*   ubf    = (u16*)carve((size_t)NROWS * DI * 2);
  float* uf     = (float*)carve((size_t)NROWS * DI * 4);
  float* xdbc   = (float*)carve((size_t)NROWS * NXC * 4);
  u16*   dtbf   = (u16*)carve((size_t)NROWS * DTR * 2);
  float* dtf    = (float*)carve((size_t)NROWS * DI * 4);
  u16*   ybf    = (u16*)carve((size_t)NROWS * DI * 2);

  // residual stream = masked input
  {
    int total = NROWS * DM;
    mamba_mask_copy<<<(total + 255) / 256, 256, 0, stream>>>(x, mk, out, total);
  }
  // pack all weights to bf16 WMMA-B layout (recomputed every call: deterministic)
  for (int l = 0; l < NDEPTH; ++l) {
    { size_t n = (size_t)DM * 2 * DI;
      mamba_pack_w<<<(unsigned)((n + 255) / 256), 256, 0, stream>>>(in_proj_w + l * n, pk_in + l * n, DM, 2 * DI); }
    { size_t n = (size_t)DI * NXC;
      mamba_pack_w<<<(unsigned)((n + 255) / 256), 256, 0, stream>>>(x_proj_w + l * n, pk_x + l * n, DI, NXC); }
    { size_t n = (size_t)DTR * DI;
      mamba_pack_w<<<(unsigned)((n + 255) / 256), 256, 0, stream>>>(dt_proj_w + l * n, pk_dt + l * n, DTR, DI); }
    { size_t n = (size_t)DI * DM;
      mamba_pack_w<<<(unsigned)((n + 255) / 256), 256, 0, stream>>>(out_proj_w + l * n, pk_out + l * n, DI, DM); }
  }

  for (int l = 0; l < NDEPTH; ++l) {
    mamba_rmsnorm_bf16<<<NROWS, 256, 0, stream>>>(out, norm_w + (size_t)l * DM, hbf);
    {   // in_proj: 2048x1024 @ 1024x4096 -> xz
      dim3 g((2 * DI + 63) / 64, NROWS / 128);
      mamba_gemm_wmma<0><<<g, 256, 0, stream>>>(hbf, pk_in + (size_t)l * DM * 2 * DI,
                                                xz, NROWS, 2 * DI, DM, nullptr, nullptr);
    }
    {
      int total = NROWS * DI;
      mamba_conv_silu<<<(total + 255) / 256, 256, 0, stream>>>(
          xz, conv_w + (size_t)l * DI * 4, conv_b + (size_t)l * DI, ubf, uf);
    }
    {   // x_proj: 2048x2048 @ 2048x96 -> xdbc
      dim3 g((NXC + 63) / 64, NROWS / 128);
      mamba_gemm_wmma<0><<<g, 256, 0, stream>>>(ubf, pk_x + (size_t)l * DI * NXC,
                                                xdbc, NROWS, NXC, DI, nullptr, nullptr);
    }
    {
      int total = NROWS * DTR;
      mamba_dtcvt<<<(total + 255) / 256, 256, 0, stream>>>(xdbc, dtbf);
    }
    {   // dt_proj: 2048x64 @ 64x2048, fused bias + softplus
      dim3 g((DI + 63) / 64, NROWS / 128);
      mamba_gemm_wmma<1><<<g, 256, 0, stream>>>(dtbf, pk_dt + (size_t)l * DTR * DI,
                                                dtf, NROWS, DI, DTR,
                                                dt_proj_b + (size_t)l * DI, nullptr);
    }
    mamba_scan<<<NB * (DI / 256), 256, 0, stream>>>(dtf, xdbc, uf, xz,
                                                    A_log + (size_t)l * DI * DSZ,
                                                    D_param + (size_t)l * DI, ybf);
    {   // out_proj: 2048x2048 @ 2048x1024, fused residual + mask (in-place)
      dim3 g((DM + 63) / 64, NROWS / 128);
      mamba_gemm_wmma<2><<<g, 256, 0, stream>>>(ybf, pk_out + (size_t)l * DI * DM,
                                                out, NROWS, DM, DI, nullptr, mk);
    }
  }
  (void)in_sizes; (void)n_in; (void)out_size; (void)ws_size;
}
